// ATLAS_8409545965987
// MI455X (gfx1250) — matrix operand order repaired and checked by reading, not compiled, not run on hardware
//
#include <hip/hip_runtime.h>
#include <hip/hip_bf16.h>

typedef __bf16 bf16;
typedef __bf16 v8bf  __attribute__((ext_vector_type(8)));
typedef __bf16 v16bf __attribute__((ext_vector_type(16)));
typedef float  v8f   __attribute__((ext_vector_type(8)));

#define B_ 32
#define T_ 128
#define S_ 256
#define E_ 512
#define H_ 1024
#define G_ 4096   // 4*H

// ---------------------------------------------------------------------------
// Generic WMMA bf16 GEMM:  Y[M,N] = X[M,K] @ W[N,K]^T (+ bias)
// X row-major (lda), W row-major [N,K] (ldb).  Optional f32 out Yf (ldcf)
// and/or bf16 out Yb (ldcb).
// Block = 128 thr = 4 waves (2 M-tiles x 2 N-groups); each wave computes a
// 16x64 tile with 4 v_wmma_f32_16x16x32_bf16 accumulators sharing one A
// fragment.  K-loop steps 64 with ping-pong fragment sets (manual 2x unroll):
// loads for slab k+32/k+64 are issued before the WMMAs of slab k/k+32, and
// land directly in the consumed registers (no rotation moves).
// Grid: (N/128, M/32).  K must be a multiple of 64.
// ---------------------------------------------------------------------------
__global__ __launch_bounds__(128)
void gemm_bf16_wmma(const bf16* __restrict__ X, int lda,
                    const bf16* __restrict__ W, int ldb,
                    const float* __restrict__ bias,
                    float* __restrict__ Yf, int ldcf,
                    bf16* __restrict__ Yb, int ldcb,
                    int K)
{
    const int lane = threadIdx.x & 31;
    const int w    = threadIdx.x >> 5;
    const int wm   = w & 1;        // M tile within block
    const int wn   = w >> 1;       // N group of 64 within block
    const int m0 = blockIdx.y * 32 + wm * 16;
    const int n0 = blockIdx.x * 128 + wn * 64;

    // A fragment addressing (16-bit A 16x32 layout):
    //   row = m0 + lane%16 ; elems 0..15 = K kk+(lane/16)*16..+15 (contiguous)
    const int arow = m0 + (lane & 15);
    const int alo  = (lane >> 4) * 16;
    // B fragment addressing (B 32x16): col = n0 + lane%16, 16 contiguous K
    const int bcol = n0 + (lane & 15);
    const int bk   = (lane >> 4) * 16;

    const bf16* xp  = X + (size_t)arow * lda + alo;          // + kk
    const bf16* wq0 = W + (size_t)(bcol)      * ldb + bk;    // + kk
    const bf16* wq1 = W + (size_t)(bcol + 16) * ldb + bk;
    const bf16* wq2 = W + (size_t)(bcol + 32) * ldb + bk;
    const bf16* wq3 = W + (size_t)(bcol + 48) * ldb + bk;

    v8f acc0 = {}, acc1 = {}, acc2 = {}, acc3 = {};

    // ping-pong fragment sets
    v8bf  aLoA, aHiA, aLoB, aHiB;
    v16bf b0A, b1A, b2A, b3A, b0B, b1B, b2B, b3B;

    // prologue: slab 0 -> set A
    aLoA = *(const v8bf*)(xp);
    aHiA = *(const v8bf*)(xp + 8);
    b0A  = *(const v16bf*)(wq0);
    b1A  = *(const v16bf*)(wq1);
    b2A  = *(const v16bf*)(wq2);
    b3A  = *(const v16bf*)(wq3);

    for (int kk = 0; kk < K; kk += 64) {
        // slab kk+32 -> set B (always valid: K % 64 == 0)
        const int k1 = kk + 32;
        aLoB = *(const v8bf*)(xp + k1);
        aHiB = *(const v8bf*)(xp + k1 + 8);
        b0B  = *(const v16bf*)(wq0 + k1);
        b1B  = *(const v16bf*)(wq1 + k1);
        b2B  = *(const v16bf*)(wq2 + k1);
        b3B  = *(const v16bf*)(wq3 + k1);

        v16bf a;
#pragma unroll
        for (int i = 0; i < 8; ++i) { a[i] = aLoA[i]; a[i + 8] = aHiA[i]; }
        acc0 = __builtin_amdgcn_wmma_f32_16x16x32_bf16(false, a, false, b0A,
                                                       (short)0, acc0, false, false);
        acc1 = __builtin_amdgcn_wmma_f32_16x16x32_bf16(false, a, false, b1A,
                                                       (short)0, acc1, false, false);
        acc2 = __builtin_amdgcn_wmma_f32_16x16x32_bf16(false, a, false, b2A,
                                                       (short)0, acc2, false, false);
        acc3 = __builtin_amdgcn_wmma_f32_16x16x32_bf16(false, a, false, b3A,
                                                       (short)0, acc3, false, false);

        const int k2 = kk + 64;
        if (k2 < K) {
            // slab kk+64 -> set A; overlaps set-B WMMAs below
            aLoA = *(const v8bf*)(xp + k2);
            aHiA = *(const v8bf*)(xp + k2 + 8);
            b0A  = *(const v16bf*)(wq0 + k2);
            b1A  = *(const v16bf*)(wq1 + k2);
            b2A  = *(const v16bf*)(wq2 + k2);
            b3A  = *(const v16bf*)(wq3 + k2);
            if (k2 + 32 < K) {  // pull slab k+96 toward cache (global_prefetch_b8)
                __builtin_prefetch((const void*)(wq0 + k2 + 32), 0, 1);
                __builtin_prefetch((const void*)(wq2 + k2 + 32), 0, 1);
                __builtin_prefetch((const void*)(xp  + k2 + 32), 0, 1);
            }
        }

        v16bf a2;
#pragma unroll
        for (int i = 0; i < 8; ++i) { a2[i] = aLoB[i]; a2[i + 8] = aHiB[i]; }
        acc0 = __builtin_amdgcn_wmma_f32_16x16x32_bf16(false, a2, false, b0B,
                                                       (short)0, acc0, false, false);
        acc1 = __builtin_amdgcn_wmma_f32_16x16x32_bf16(false, a2, false, b1B,
                                                       (short)0, acc1, false, false);
        acc2 = __builtin_amdgcn_wmma_f32_16x16x32_bf16(false, a2, false, b2B,
                                                       (short)0, acc2, false, false);
        acc3 = __builtin_amdgcn_wmma_f32_16x16x32_bf16(false, a2, false, b3B,
                                                       (short)0, acc3, false, false);
    }

    // C/D layout: m = m0 + r + 8*(lane/16), n = n0 + lane%16 (+16/+32/+48)
    const int mrow = m0 + (lane >> 4) * 8;
    const int ncol = n0 + (lane & 15);
#pragma unroll
    for (int r = 0; r < 8; ++r) {
        float v0 = acc0[r], v1 = acc1[r], v2 = acc2[r], v3 = acc3[r];
        if (bias) {
            v0 += bias[ncol];      v1 += bias[ncol + 16];
            v2 += bias[ncol + 32]; v3 += bias[ncol + 48];
        }
        const int m = mrow + r;
        if (Yf) {
            Yf[(size_t)m * ldcf + ncol]      = v0;
            Yf[(size_t)m * ldcf + ncol + 16] = v1;
            Yf[(size_t)m * ldcf + ncol + 32] = v2;
            Yf[(size_t)m * ldcf + ncol + 48] = v3;
        }
        if (Yb) {
            Yb[(size_t)m * ldcb + ncol]      = (bf16)v0;
            Yb[(size_t)m * ldcb + ncol + 16] = (bf16)v1;
            Yb[(size_t)m * ldcb + ncol + 32] = (bf16)v2;
            Yb[(size_t)m * ldcb + ncol + 48] = (bf16)v3;
        }
    }
}

// ---------------------------------------------------------------------------
// Attention: scores = q.K / sqrt(H) (masked) -> softmax -> av = attn.V (bf16)
// One block per batch element, 256 threads (8 waves).
// ---------------------------------------------------------------------------
__global__ __launch_bounds__(256)
void attn_kernel(const float* __restrict__ q,            // [B,H]
                 const float* __restrict__ Kp,           // [S,B,H]
                 const float* __restrict__ Vp,           // [S,B,H]
                 const unsigned char* __restrict__ mask, // [B,S] (bool)
                 bf16* __restrict__ av)                  // [B,H]
{
    __shared__ float qs[H_];
    __shared__ float sc[S_];
    __shared__ float red[8];
    const int b    = blockIdx.x;
    const int tid  = threadIdx.x;
    const int lane = tid & 31;
    const int w    = tid >> 5;

    for (int h = tid; h < H_; h += 256) qs[h] = q[b * H_ + h];
    __syncthreads();

    const float scale = 0.03125f; // 1/sqrt(1024)
    for (int it = 0; it < S_ / 8; ++it) {
        const int s = it * 8 + w;
        const float* kr = Kp + ((size_t)s * B_ + b) * H_;
        float acc = 0.f;
#pragma unroll 4
        for (int j = 0; j < H_ / 32; ++j)
            acc += qs[lane + 32 * j] * kr[lane + 32 * j];
#pragma unroll
        for (int off = 16; off > 0; off >>= 1)
            acc += __shfl_xor(acc, off, 32);
        if (lane == 0)
            sc[s] = mask[b * S_ + s] ? -1e9f : acc * scale;
    }
    __syncthreads();

    // softmax over 256 scores; thread owns sc[tid]
    const float v = sc[tid];
    float m = v;
#pragma unroll
    for (int off = 16; off > 0; off >>= 1) m = fmaxf(m, __shfl_xor(m, off, 32));
    if (lane == 0) red[w] = m;
    __syncthreads();
    m = red[0];
#pragma unroll
    for (int i = 1; i < 8; ++i) m = fmaxf(m, red[i]);
    const float e = __expf(v - m);
    float ssum = e;
#pragma unroll
    for (int off = 16; off > 0; off >>= 1) ssum += __shfl_xor(ssum, off, 32);
    __syncthreads();
    if (lane == 0) red[w] = ssum;
    __syncthreads();
    float tot = 0.f;
#pragma unroll
    for (int i = 0; i < 8; ++i) tot += red[i];
    sc[tid] = e / tot;
    __syncthreads();

    // ctx = sum_s attn[s] * V[s,b,:]  (threads sweep h, coalesced over V rows)
    float a0 = 0.f, a1 = 0.f, a2 = 0.f, a3 = 0.f;
    for (int s = 0; s < S_; ++s) {
        const float a = sc[s];
        const float* vr = Vp + ((size_t)s * B_ + b) * H_;
        a0 += a * vr[tid];
        a1 += a * vr[tid + 256];
        a2 += a * vr[tid + 512];
        a3 += a * vr[tid + 768];
    }
    av[b * H_ + tid]       = (bf16)a0;
    av[b * H_ + tid + 256] = (bf16)a1;
    av[b * H_ + tid + 512] = (bf16)a2;
    av[b * H_ + tid + 768] = (bf16)a3;
}

// ---------------------------------------------------------------------------
// LSTM elementwise cells
// ---------------------------------------------------------------------------
__device__ __forceinline__ float sigmoidf_(float x) { return 1.f / (1.f + __expf(-x)); }

__global__ __launch_bounds__(256)
void cell0_kernel(const float* __restrict__ g, float* __restrict__ c0,
                  bf16* __restrict__ xcat0, bf16* __restrict__ xcat1)
{
    const int idx = blockIdx.x * 256 + threadIdx.x; // B*H
    const int b = idx >> 10, h = idx & 1023;
    const float* gr = g + (size_t)b * G_;
    const float i  = sigmoidf_(gr[h]);
    const float f  = sigmoidf_(gr[h + 1024]);
    const float gg = tanhf(gr[h + 2048]);
    const float o  = sigmoidf_(gr[h + 3072]);
    const float c  = f * c0[idx] + i * gg;
    c0[idx] = c;
    const float hh = o * tanhf(c);
    xcat0[(size_t)b * 2560 + 1536 + h] = (bf16)hh; // h0_prev slot for next step
    xcat1[(size_t)b * 2048 + h]        = (bf16)hh; // input to layer-1 GEMM now
}

__global__ __launch_bounds__(256)
void cell1_kernel(const float* __restrict__ g, float* __restrict__ c1,
                  bf16* __restrict__ xcat1, bf16* __restrict__ c1b,
                  float* __restrict__ out, int t)
{
    const int idx = blockIdx.x * 256 + threadIdx.x; // B*H
    const int b = idx >> 10, h = idx & 1023;
    const float* gr = g + (size_t)b * G_;
    const float i  = sigmoidf_(gr[h]);
    const float f  = sigmoidf_(gr[h + 1024]);
    const float gg = tanhf(gr[h + 2048]);
    const float o  = sigmoidf_(gr[h + 3072]);
    const float c  = f * c1[idx] + i * gg;
    c1[idx] = c;
    const float hh = o * tanhf(c);
    out[((size_t)b * T_ + t) * H_ + h]  = hh;       // decoder output
    xcat1[(size_t)b * 2048 + 1024 + h]  = (bf16)hh; // h1_prev for next step
    c1b[idx]                            = (bf16)c;  // query source next step
}

// ---------------------------------------------------------------------------
// Setup / glue kernels
// ---------------------------------------------------------------------------
__global__ void f2b_kernel(const float* __restrict__ s, bf16* __restrict__ d, int n)
{
    const int i = blockIdx.x * 256 + threadIdx.x;
    if (i < n) d[i] = (bf16)s[i];
}

__global__ void cat2_kernel(const float* __restrict__ A, int Ka,
                            const float* __restrict__ Bm, int Kb,
                            bf16* __restrict__ d, long total)
{
    const long i = (long)blockIdx.x * 256 + threadIdx.x;
    if (i < total) {
        const int Kc = Ka + Kb;
        const long n = i / Kc;
        const int  k = (int)(i % Kc);
        d[i] = (bf16)(k < Ka ? A[n * Ka + k] : Bm[n * Kb + (k - Ka)]);
    }
}

__global__ void badd_kernel(const float* __restrict__ a, const float* __restrict__ b,
                            float* __restrict__ d, int n)
{
    const int i = blockIdx.x * 256 + threadIdx.x;
    if (i < n) d[i] = a[i] + b[i];
}

__global__ void init_kernel(const float* __restrict__ enc_c,  // [2,B,H/2]
                            float* __restrict__ c0f, float* __restrict__ c1f,
                            bf16* __restrict__ c1b,
                            bf16* __restrict__ xcat0, bf16* __restrict__ xcat1)
{
    const int idx = blockIdx.x * 256 + threadIdx.x; // B*H
    const int b = idx >> 10, h = idx & 1023;
    const int d = h >> 9, j = h & 511;
    const float val = enc_c[((size_t)d * B_ + b) * 512 + j];
    c0f[idx] = val;
    c1f[idx] = val;
    c1b[idx] = (bf16)val;
    xcat0[(size_t)b * 2560 + 1536 + h] = (bf16)0.f; // h0_prev = 0
    xcat1[(size_t)b * 2048 + 1024 + h] = (bf16)0.f; // h1_prev = 0
}

__global__ void xcopy_kernel(const bf16* __restrict__ xbf, bf16* __restrict__ xcat0, int t)
{
    const int idx = blockIdx.x * 256 + threadIdx.x; // B*E
    const int b = idx >> 9, e = idx & 511;
    xcat0[(size_t)b * 2560 + 1024 + e] = xbf[((size_t)b * T_ + t) * E_ + e];
}

// ---------------------------------------------------------------------------
extern "C" void kernel_launch(void* const* d_in, const int* in_sizes, int n_in,
                              void* d_out, int out_size, void* d_ws, size_t ws_size,
                              hipStream_t stream)
{
    const float* inputs  = (const float*)d_in[0];   // [B,T,E]
    const float* enc_out = (const float*)d_in[1];   // [S,B,H]
    const float* enc_c   = (const float*)d_in[2];   // [2,B,H/2]
    const unsigned char* mask = (const unsigned char*)d_in[3]; // [B,S]
    const float* Wq = (const float*)d_in[4];  const float* bq = (const float*)d_in[5];
    const float* Wk = (const float*)d_in[6];  const float* bk = (const float*)d_in[7];
    const float* Wv = (const float*)d_in[8];  const float* bv = (const float*)d_in[9];
    const float* Wo = (const float*)d_in[10]; const float* bo = (const float*)d_in[11];
    const float* Wih0 = (const float*)d_in[12]; const float* Whh0 = (const float*)d_in[13];
    const float* bih0 = (const float*)d_in[14]; const float* bhh0 = (const float*)d_in[15];
    const float* Wih1 = (const float*)d_in[16]; const float* Whh1 = (const float*)d_in[17];
    const float* bih1 = (const float*)d_in[18]; const float* bhh1 = (const float*)d_in[19];
    float* out = (float*)d_out;                      // [B,T,H]

    // -- workspace bump allocator (256B aligned) --
    size_t off = 0;
    auto alloc = [&](size_t bytes) -> char* {
        char* p = (char*)d_ws + off;
        off += (bytes + 255) & ~(size_t)255;
        return p;
    };
    bf16* Wq_b  = (bf16*)alloc((size_t)H_ * H_ * 2);
    bf16* Wk_b  = (bf16*)alloc((size_t)H_ * H_ * 2);
    bf16* Wv_b  = (bf16*)alloc((size_t)H_ * H_ * 2);
    bf16* Wo_b  = (bf16*)alloc((size_t)H_ * H_ * 2);
    bf16* Wc0   = (bf16*)alloc((size_t)G_ * 2560 * 2);   // [Wih0 | Whh0]
    bf16* Wc1   = (bf16*)alloc((size_t)G_ * 2048 * 2);   // [Wih1 | Whh1]
    bf16* enc_b = (bf16*)alloc((size_t)S_ * B_ * H_ * 2);
    bf16* x_b   = (bf16*)alloc((size_t)B_ * T_ * E_ * 2);
    float* Kf   = (float*)alloc((size_t)S_ * B_ * H_ * 4);
    float* Vf   = (float*)alloc((size_t)S_ * B_ * H_ * 4);
    float* qf   = (float*)alloc((size_t)B_ * H_ * 4);
    float* g0   = (float*)alloc((size_t)B_ * G_ * 4);
    float* g1   = (float*)alloc((size_t)B_ * G_ * 4);
    float* c0f  = (float*)alloc((size_t)B_ * H_ * 4);
    float* c1f  = (float*)alloc((size_t)B_ * H_ * 4);
    float* bg0  = (float*)alloc((size_t)G_ * 4);
    float* bg1  = (float*)alloc((size_t)G_ * 4);
    bf16* xcat0 = (bf16*)alloc((size_t)B_ * 2560 * 2);   // [ctx | x_t | h0_prev]
    bf16* xcat1 = (bf16*)alloc((size_t)B_ * 2048 * 2);   // [h0 | h1_prev]
    bf16* c1b   = (bf16*)alloc((size_t)B_ * H_ * 2);
    bf16* avb   = (bf16*)alloc((size_t)B_ * H_ * 2);
    (void)ws_size; (void)n_in; (void)in_sizes; (void)out_size;

    // -- one-time setup --
    const int HH = H_ * H_;
    f2b_kernel<<<(HH + 255) / 256, 256, 0, stream>>>(Wq, Wq_b, HH);
    f2b_kernel<<<(HH + 255) / 256, 256, 0, stream>>>(Wk, Wk_b, HH);
    f2b_kernel<<<(HH + 255) / 256, 256, 0, stream>>>(Wv, Wv_b, HH);
    f2b_kernel<<<(HH + 255) / 256, 256, 0, stream>>>(Wo, Wo_b, HH);
    const int NE = S_ * B_ * H_;
    f2b_kernel<<<(NE + 255) / 256, 256, 0, stream>>>(enc_out, enc_b, NE);
    const int NX = B_ * T_ * E_;
    f2b_kernel<<<(NX + 255) / 256, 256, 0, stream>>>(inputs, x_b, NX);
    {
        long tot0 = (long)G_ * 2560;
        cat2_kernel<<<(unsigned)((tot0 + 255) / 256), 256, 0, stream>>>(Wih0, 1536, Whh0, 1024, Wc0, tot0);
        long tot1 = (long)G_ * 2048;
        cat2_kernel<<<(unsigned)((tot1 + 255) / 256), 256, 0, stream>>>(Wih1, 1024, Whh1, 1024, Wc1, tot1);
    }
    badd_kernel<<<(G_ + 255) / 256, 256, 0, stream>>>(bih0, bhh0, bg0, G_);
    badd_kernel<<<(G_ + 255) / 256, 256, 0, stream>>>(bih1, bhh1, bg1, G_);
    init_kernel<<<(B_ * H_) / 256, 256, 0, stream>>>(enc_c, c0f, c1f, c1b, xcat0, xcat1);

    // K/V projections: [S*B, H] @ W^T  (one-time 34 GFLOP, WMMA)
    {
        dim3 grid(H_ / 128, (S_ * B_) / 32);
        gemm_bf16_wmma<<<grid, 128, 0, stream>>>(enc_b, H_, Wk_b, H_, bk, Kf, H_, (bf16*)nullptr, 0, H_);
        gemm_bf16_wmma<<<grid, 128, 0, stream>>>(enc_b, H_, Wv_b, H_, bv, Vf, H_, (bf16*)nullptr, 0, H_);
    }

    // -- 128 sequential decode steps --
    for (int t = 0; t < T_; ++t) {
        // q = c1 @ Wq^T + bq
        gemm_bf16_wmma<<<dim3(H_ / 128, 1), 128, 0, stream>>>(
            c1b, H_, Wq_b, H_, bq, qf, H_, (bf16*)nullptr, 0, H_);
        // attention -> av (bf16)
        attn_kernel<<<B_, 256, 0, stream>>>(qf, Kf, Vf, mask, avb);
        // ctx = av @ Wo^T + bo -> xcat0[:, 0:1024] (bf16)
        gemm_bf16_wmma<<<dim3(H_ / 128, 1), 128, 0, stream>>>(
            avb, H_, Wo_b, H_, bo, (float*)nullptr, 0, xcat0, 2560, H_);
        // x_t -> xcat0[:, 1024:1536]
        xcopy_kernel<<<(B_ * E_) / 256, 256, 0, stream>>>(x_b, xcat0, t);
        // gates0 = [ctx|x_t|h0_prev] @ [Wih0|Whh0]^T + (bih0+bhh0)
        gemm_bf16_wmma<<<dim3(G_ / 128, 1), 128, 0, stream>>>(
            xcat0, 2560, Wc0, 2560, bg0, g0, G_, (bf16*)nullptr, 0, 2560);
        cell0_kernel<<<(B_ * H_) / 256, 256, 0, stream>>>(g0, c0f, xcat0, xcat1);
        // gates1 = [h0|h1_prev] @ [Wih1|Whh1]^T + (bih1+bhh1)
        gemm_bf16_wmma<<<dim3(G_ / 128, 1), 128, 0, stream>>>(
            xcat1, 2048, Wc1, 2048, bg1, g1, G_, (bf16*)nullptr, 0, 2048);
        cell1_kernel<<<(B_ * H_) / 256, 256, 0, stream>>>(g1, c1f, xcat1, c1b, out, t);
    }
}
